// EnhancedGraphMambaModel_71992241815701
// MI455X (gfx1250) — matrix operand order repaired and checked by reading, not compiled, not run on hardware
//
#include <hip/hip_runtime.h>

typedef unsigned short u16;
typedef __attribute__((ext_vector_type(16))) __bf16 v16bf;
typedef __attribute__((ext_vector_type(8)))  float  v8f;
typedef __attribute__((ext_vector_type(4)))  unsigned int u32x4;

constexpr int cB = 4, cT = 64, cN = 256, cF = 32, cD = 256;
constexpr int cDS = 16, cDC = 4, cDI = 512, cDR = 16, cH = 4, cE = 2048;
constexpr int cG  = cB * cT;      // 256 graphs
constexpr int cGN = cG * cN;      // 65536 rows
constexpr float cEPS = 1e-5f;

__device__ __forceinline__ u16 f2bf(float f) {
  unsigned u = __float_as_uint(f);
  unsigned r = u + 0x7FFFu + ((u >> 16) & 1u);   // round-to-nearest-even
  return (u16)(r >> 16);
}
__device__ __forceinline__ float bf2f(u16 h) { return __uint_as_float(((unsigned)h) << 16); }
__device__ __forceinline__ float siluf(float x) { return x / (1.f + __expf(-x)); }
__device__ __forceinline__ float softplusf(float x) { return (x > 20.f) ? x : log1pf(__expf(x)); }

// ---------------- graph preprocessing ----------------
__global__ void k_graph_init(float* deg, int* cnt, int* cur) {
  int n = threadIdx.x; deg[n] = 0.f; cnt[n] = 0; cur[n] = 0;
}
__global__ void k_deg(const int* __restrict__ ei, float* deg, int* cnt) {
  int e = blockIdx.x * 256 + threadIdx.x;
  if (e >= cE + cN) return;
  int dst = (e < cE) ? ei[cE + e] : (e - cE);
  atomicAdd(&deg[dst], 1.f);
  atomicAdd(&cnt[dst], 1);
}
__global__ void k_off_dinv(const int* __restrict__ cnt, int* off,
                           const float* __restrict__ deg, float* dinv) {
  int n = threadIdx.x;
  dinv[n] = rsqrtf(fmaxf(deg[n], 1.f));
  if (n == 0) {
    int a = 0;
    for (int i = 0; i < cN; i++) { off[i] = a; a += cnt[i]; }
    off[cN] = a;
  }
}
__global__ void k_fill(const int* __restrict__ ei, const int* __restrict__ off, int* cur,
                       const float* __restrict__ dinv, int* colsrc, float* colw) {
  int e = blockIdx.x * 256 + threadIdx.x;
  if (e >= cE + cN) return;
  int src = (e < cE) ? ei[e] : (e - cE);
  int dst = (e < cE) ? ei[cE + e] : (e - cE);
  int pos = atomicAdd(&cur[dst], 1);
  int idx = off[dst] + pos;
  colsrc[idx] = src;
  colw[idx] = dinv[src] * dinv[dst];
}

// ---------------- input layernorm (F=32) -> bf16 ----------------
__global__ void k_input_ln(const float* __restrict__ snap, const float* __restrict__ g,
                           const float* __restrict__ b, u16* __restrict__ xbf) {
  int row = blockIdx.x * 256 + threadIdx.x;
  if (row >= cGN) return;
  const float* x = snap + (size_t)row * cF;
  float s = 0.f, s2 = 0.f;
#pragma unroll
  for (int i = 0; i < cF; i++) { float v = x[i]; s += v; s2 += v * v; }
  float mu = s * (1.f / cF);
  float var = s2 * (1.f / cF) - mu * mu;
  float inv = rsqrtf(var + cEPS);
  u16* o = xbf + (size_t)row * cF;
#pragma unroll
  for (int i = 0; i < cF; i++) o[i] = f2bf((x[i] - mu) * inv * g[i] + b[i]);
}

// ---------------- weight convert f32[K x Nc] -> bf16 transposed [Nc x K] ----------------
__global__ void k_cvtT(const float* __restrict__ W, u16* __restrict__ Wt, int K, int Nc) {
  int i = blockIdx.x * 256 + threadIdx.x;
  if (i >= K * Nc) return;
  int k = i / Nc, n = i - k * Nc;
  Wt[(size_t)n * K + k] = f2bf(W[i]);
}

// ---------------- bf16 WMMA GEMM: C[MxNc] = A[MxK] @ W[KxNc]  (Wt is Nc x K) ----------------
// Each wave computes a 16 x (16*NT) strip: one A fragment reused across NT B tiles.
template <int NT>
__global__ __launch_bounds__(32) void k_gemmN(const u16* __restrict__ A, const u16* __restrict__ Wt,
                                              float* __restrict__ C, int M, int K, int Nc) {
  int nstrips = Nc / (16 * NT);
  int tile = blockIdx.x;
  int tm = tile / nstrips, ts = tile - tm * nstrips;
  int lane = threadIdx.x;
  int half = lane >> 4;       // 0 | 1
  int l15  = lane & 15;
  // A fragment: lanes 0-15 row M=l15 hold K 0..7 & 16..23; lanes 16-31 hold K 8..15 & 24..31
  const u16* arow  = A  + (size_t)(tm * 16 + l15) * K + half * 8;
  // B fragment: col N = l15 (+16*nt); lanes 0-15 K 0..7 & 16..23; lanes 16-31 K 8..15 & 24..31
  const u16* bbase = Wt + (size_t)(ts * NT * 16 + l15) * K + half * 8;
  v8f acc[NT] = {};
  for (int kk = 0; kk < K; kk += 32) {
    union { u32x4 q[2]; v16bf v; } a;
    a.q[0] = *(const u32x4*)(arow + kk);
    a.q[1] = *(const u32x4*)(arow + kk + 16);
    __builtin_prefetch(arow + kk + 64, 0, 0);   // global_prefetch_b8: next A chunk
#pragma unroll
    for (int nt = 0; nt < NT; ++nt) {
      union { u32x4 q[2]; v16bf v; } b;
      const u16* bcol = bbase + (size_t)nt * 16 * K;
      b.q[0] = *(const u32x4*)(bcol + kk);
      b.q[1] = *(const u32x4*)(bcol + kk + 16);
      acc[nt] = __builtin_amdgcn_wmma_f32_16x16x32_bf16(false, a.v, false, b.v, (short)0,
                                                        acc[nt], false, false);
    }
  }
  // C/D: VGPR v -> row M = v + 8*half, col N = l15
#pragma unroll
  for (int nt = 0; nt < NT; ++nt) {
    float* crow = C + (size_t)(tm * 16 + half * 8) * Nc + (ts * NT + nt) * 16 + l15;
#pragma unroll
    for (int v = 0; v < 8; ++v) crow[(size_t)v * Nc] = acc[nt][v];
  }
}

// ---------------- GCN aggregate: o = hw_r + gb + rb + sum_e norm*hw_g[src] ----------------
__global__ void k_aggregate(const float* __restrict__ hwg, float* __restrict__ o,
                            const float* __restrict__ gb, const float* __restrict__ rb,
                            const int* __restrict__ off, const int* __restrict__ colsrc,
                            const float* __restrict__ colw) {
  int gn = blockIdx.x;
  int g = gn >> 8, n = gn & 255;
  int d = threadIdx.x;
  float acc = o[(size_t)gn * cD + d] + gb[d] + rb[d];
  const float* base = hwg + (size_t)g * cN * cD;
  int s0 = off[n], s1 = off[n + 1];
  for (int j = s0; j < s1; j++)
    acc += colw[j] * base[(size_t)colsrc[j] * cD + d];
  o[(size_t)gn * cD + d] = acc;
}

// ---------------- LayerNorm over 256 (+ optional residual/bias/SiLU) ----------------
__global__ void k_ln256(const float* __restrict__ x, const float* __restrict__ res,
                        const float* __restrict__ resb, const float* __restrict__ g,
                        const float* __restrict__ b, u16* __restrict__ obf,
                        float* __restrict__ of32, int act) {
  __shared__ float s1[256], s2[256];
  int row = blockIdx.x, d = threadIdx.x;
  size_t idx = (size_t)row * 256 + d;
  float v = x[idx];
  if (res)  v += res[idx];
  if (resb) v += resb[d];
  s1[d] = v; s2[d] = v * v;
  __syncthreads();
  for (int o2 = 128; o2 > 0; o2 >>= 1) {
    if (d < o2) { s1[d] += s1[d + o2]; s2[d] += s2[d + o2]; }
    __syncthreads();
  }
  float mu = s1[0] * (1.f / 256.f);
  float var = s2[0] * (1.f / 256.f) - mu * mu;
  float inv = rsqrtf(var + cEPS);
  float o = (v - mu) * inv * g[d] + b[d];
  if (act) o = siluf(o);
  if (obf)  obf[idx]  = f2bf(o);
  if (of32) of32[idx] = o;
}

// ---------------- mean pool over N -> emb (f32 + bf16) ----------------
__global__ void k_pool(const u16* __restrict__ x, float* __restrict__ emb, u16* __restrict__ embbf) {
  int g = blockIdx.x, d = threadIdx.x;
  const u16* base = x + (size_t)g * cN * cD + d;
  float acc = 0.f;
  for (int n = 0; n < cN; n++) acc += bf2f(base[(size_t)n * cD]);
  acc *= (1.f / cN);
  emb[g * cD + d] = acc;
  embbf[g * cD + d] = f2bf(acc);
}

// ---------------- attention (per (b,h) block, online softmax) ----------------
__global__ void k_attn(const float* __restrict__ qkv, const float* __restrict__ bias,
                       u16* __restrict__ aobf) {
  int bh = blockIdx.x;
  int b = bh / cH, h = bh % cH;
  int q = threadIdx.x;                 // t index, 64 threads
  int grow = b * cT + q;
  float qv[64], out[64];
#pragma unroll
  for (int d0 = 0; d0 < 64; ++d0) {
    qv[d0] = qkv[(size_t)grow * 768 + h * 64 + d0] + bias[h * 64 + d0];
    out[d0] = 0.f;
  }
  float m = -1e30f, den = 0.f;
  for (int k = 0; k < cT; k++) {
    size_t krow = (size_t)(b * cT + k) * 768;
    float s = 0.f;
#pragma unroll
    for (int d0 = 0; d0 < 64; ++d0)
      s += qv[d0] * (qkv[krow + 256 + h * 64 + d0] + bias[256 + h * 64 + d0]);
    s *= 0.125f;                       // 1/sqrt(64)
    float mn = fmaxf(m, s);
    float sc = __expf(m - mn);
    float w  = __expf(s - mn);
    den = den * sc + w;
#pragma unroll
    for (int d0 = 0; d0 < 64; ++d0)
      out[d0] = out[d0] * sc + w * (qkv[krow + 512 + h * 64 + d0] + bias[512 + h * 64 + d0]);
    m = mn;
  }
  float r = 1.f / den;
#pragma unroll
  for (int d0 = 0; d0 < 64; ++d0)
    aobf[(size_t)grow * cD + h * 64 + d0] = f2bf(out[d0] * r);
}

// ---------------- causal conv (width 4) + SiLU ----------------
__global__ void k_conv(const float* __restrict__ xz, const float* __restrict__ cw,
                       const float* __restrict__ cb, float* __restrict__ us,
                       u16* __restrict__ usbf) {
  int idx = blockIdx.x * 256 + threadIdx.x;
  if (idx >= cB * cDI) return;
  int b = idx / cDI, i = idx - (idx / cDI) * cDI;
  float w0 = cw[i * 4 + 0], w1 = cw[i * 4 + 1], w2 = cw[i * 4 + 2], w3 = cw[i * 4 + 3];
  float bias = cb[i];
  float u0 = 0.f, u1 = 0.f, u2 = 0.f;
  for (int t = 0; t < cT; t++) {
    size_t row = (size_t)(b * cT + t);
    float ut = xz[row * 1024 + i];
    float c = bias + w0 * u0 + w1 * u1 + w2 * u2 + w3 * ut;
    float o = siluf(c);
    us[row * cDI + i] = o;
    usbf[row * cDI + i] = f2bf(o);
    u0 = u1; u1 = u2; u2 = ut;
  }
}

// ---------------- dt = softplus(dtr @ m_dt_w + b) ----------------
__global__ void k_dt(const float* __restrict__ xdb, const float* __restrict__ dtw,
                     const float* __restrict__ dtb, float* __restrict__ dt) {
  int idx = blockIdx.x * 256 + threadIdx.x;
  if (idx >= cG * cDI) return;
  int g = idx / cDI, i = idx - g * cDI;
  float a = dtb[i];
#pragma unroll
  for (int r = 0; r < cDR; r++) a += xdb[(size_t)g * 48 + r] * dtw[r * cDI + i];
  dt[idx] = softplusf(a);
}

// ---------------- selective scan; only t = T-1 output kept ----------------
__global__ void k_scan(const float* __restrict__ dt, const float* __restrict__ us,
                       const float* __restrict__ xdb, const float* __restrict__ xz,
                       const float* __restrict__ Alog, const float* __restrict__ mD,
                       float* __restrict__ ylast) {
  int idx = blockIdx.x * 256 + threadIdx.x;
  if (idx >= cB * cDI) return;
  int b = idx / cDI, i = idx - b * cDI;
  float A[cDS], h[cDS];
#pragma unroll
  for (int s = 0; s < cDS; s++) { A[s] = -__expf(Alog[i * cDS + s]); h[s] = 0.f; }
  float dval = mD[i];
  float yy = 0.f;
  for (int t = 0; t < cT; t++) {
    size_t g = (size_t)(b * cT + t);
    float dtv = dt[g * cDI + i];
    float uv  = us[g * cDI + i];
    float du  = dtv * uv;
    float y = 0.f;
#pragma unroll
    for (int s = 0; s < cDS; s++) {
      float dA = __expf(dtv * A[s]);
      h[s] = dA * h[s] + du * xdb[g * 48 + 16 + s];
      y += h[s] * xdb[g * 48 + 32 + s];
    }
    if (t == cT - 1) {
      float zg = xz[g * 1024 + cDI + i];
      yy = (y + uv * dval) * siluf(zg);
    }
  }
  ylast[idx] = yy;
}

// ---------------- head MLP (final timestep only) ----------------
__global__ void k_head(const float* __restrict__ yl, const float* __restrict__ moW,
                       const float* __restrict__ h1w, const float* __restrict__ h1b,
                       const float* __restrict__ l1g, const float* __restrict__ l1b,
                       const float* __restrict__ h2w, const float* __restrict__ h2b,
                       const float* __restrict__ l2g, const float* __restrict__ l2b,
                       const float* __restrict__ h3w, const float* __restrict__ h3b,
                       float* __restrict__ out) {
  __shared__ float fin[256]; __shared__ float t1[128]; __shared__ float t2[64];
  __shared__ float red[2];
  int b = blockIdx.x, d = threadIdx.x;
  float acc = 0.f;
  for (int i = 0; i < cDI; i++) acc += yl[b * cDI + i] * moW[(size_t)i * cD + d];
  fin[d] = acc; __syncthreads();
  if (d < 128) { float a = h1b[d]; for (int j = 0; j < 256; j++) a += fin[j] * h1w[j * 128 + d]; t1[d] = a; }
  __syncthreads();
  if (d == 0) {
    float s = 0.f, s2 = 0.f;
    for (int j = 0; j < 128; j++) { s += t1[j]; s2 += t1[j] * t1[j]; }
    float mu = s / 128.f; red[0] = mu; red[1] = rsqrtf(s2 / 128.f - mu * mu + cEPS);
  }
  __syncthreads();
  if (d < 128) { float o = (t1[d] - red[0]) * red[1] * l1g[d] + l1b[d]; t1[d] = siluf(o); }
  __syncthreads();
  if (d < 64) { float a = h2b[d]; for (int j = 0; j < 128; j++) a += t1[j] * h2w[j * 64 + d]; t2[d] = a; }
  __syncthreads();
  if (d == 0) {
    float s = 0.f, s2 = 0.f;
    for (int j = 0; j < 64; j++) { s += t2[j]; s2 += t2[j] * t2[j]; }
    float mu = s / 64.f; red[0] = mu; red[1] = rsqrtf(s2 / 64.f - mu * mu + cEPS);
  }
  __syncthreads();
  if (d < 64) { float o = (t2[d] - red[0]) * red[1] * l2g[d] + l2b[d]; t2[d] = siluf(o); }
  __syncthreads();
  if (d < 2) {
    float a = h3b[d];
    for (int j = 0; j < 64; j++) a += t2[j] * h3w[j * 2 + d];
    out[b * 2 + d] = softplusf(a) + 1e-6f;
  }
}

// ---------------- host launch ----------------
extern "C" void kernel_launch(void* const* d_in, const int* in_sizes, int n_in,
                              void* d_out, int out_size, void* d_ws, size_t ws_size,
                              hipStream_t stream) {
  (void)in_sizes; (void)n_in; (void)out_size; (void)ws_size;
  const float* snap  = (const float*)d_in[0];
  const int*   ei    = (const int*)d_in[1];
  const float* normg = (const float*)d_in[2];  const float* normb = (const float*)d_in[3];
  const float* g1w = (const float*)d_in[4];    const float* g1b = (const float*)d_in[5];
  const float* g2w = (const float*)d_in[6];    const float* g2b = (const float*)d_in[7];
  const float* g3w = (const float*)d_in[8];    const float* g3b = (const float*)d_in[9];
  const float* r1w = (const float*)d_in[10];   const float* r1b = (const float*)d_in[11];
  const float* r2w = (const float*)d_in[12];   const float* r2b = (const float*)d_in[13];
  const float* r3w = (const float*)d_in[14];   const float* r3b = (const float*)d_in[15];
  const float* ln1g = (const float*)d_in[16];  const float* ln1b = (const float*)d_in[17];
  const float* ln2g = (const float*)d_in[18];  const float* ln2b = (const float*)d_in[19];
  const float* ln3g = (const float*)d_in[20];  const float* ln3b = (const float*)d_in[21];
  const float* aiw = (const float*)d_in[22];   const float* aib = (const float*)d_in[23];
  const float* aow = (const float*)d_in[24];   const float* aob = (const float*)d_in[25];
  const float* alg = (const float*)d_in[26];   const float* alb = (const float*)d_in[27];
  const float* minw = (const float*)d_in[28];
  const float* mcw  = (const float*)d_in[29];  const float* mcb  = (const float*)d_in[30];
  const float* mxw  = (const float*)d_in[31];
  const float* mdtw = (const float*)d_in[32];  const float* mdtb = (const float*)d_in[33];
  const float* mAlog = (const float*)d_in[34]; const float* mDp  = (const float*)d_in[35];
  const float* mow  = (const float*)d_in[36];
  const float* h1w = (const float*)d_in[37];   const float* h1b = (const float*)d_in[38];
  const float* hl1g = (const float*)d_in[39];  const float* hl1b = (const float*)d_in[40];
  const float* h2w = (const float*)d_in[41];   const float* h2b = (const float*)d_in[42];
  const float* hl2g = (const float*)d_in[43];  const float* hl2b = (const float*)d_in[44];
  const float* h3w = (const float*)d_in[45];   const float* h3b = (const float*)d_in[46];
  float* out = (float*)d_out;

  char* p = (char*)d_ws;
  auto carve = [&](size_t bytes) -> void* {
    void* r = (void*)p;
    p += (bytes + 255) & ~(size_t)255;
    return r;
  };
  float* wDeg  = (float*)carve(cN * 4);
  int*   wCnt  = (int*)carve(cN * 4);
  int*   wOff  = (int*)carve((cN + 1) * 4);
  int*   wCur  = (int*)carve(cN * 4);
  float* wDinv = (float*)carve(cN * 4);
  int*   wColS = (int*)carve((cE + cN) * 4);
  float* wColW = (float*)carve((cE + cN) * 4);
  u16*   wWt   = (u16*)carve((size_t)256 * 1024 * 2);     // largest transposed weight
  u16*   wX0   = (u16*)carve((size_t)cGN * cD * 2);       // 32 MB
  u16*   wX1   = (u16*)carve((size_t)cGN * cD * 2);       // 32 MB
  float* wHwG  = (float*)carve((size_t)cGN * cD * 4);     // 64 MB
  float* wHwR  = (float*)carve((size_t)cGN * cD * 4);     // 64 MB
  float* wEmb  = (float*)carve((size_t)cG * cD * 4);
  u16*   wEmbBf = (u16*)carve((size_t)cG * cD * 2);
  float* wQkv  = (float*)carve((size_t)cG * 3 * cD * 4);
  u16*   wAoBf = (u16*)carve((size_t)cG * cD * 2);
  float* wAoPr = (float*)carve((size_t)cG * cD * 4);
  u16*   wEmb2Bf = (u16*)carve((size_t)cG * cD * 2);
  float* wXz   = (float*)carve((size_t)cG * 2 * cDI * 4);
  float* wUs   = (float*)carve((size_t)cG * cDI * 4);
  u16*   wUsBf = (u16*)carve((size_t)cG * cDI * 2);
  float* wXdb  = (float*)carve((size_t)cG * 48 * 4);
  float* wDt   = (float*)carve((size_t)cG * cDI * 4);
  float* wYl   = (float*)carve((size_t)cB * cDI * 4);

  auto cvt = [&](const float* W, int K, int Nc) {
    int tot = K * Nc;
    k_cvtT<<<(tot + 255) / 256, 256, 0, stream>>>(W, wWt, K, Nc);
  };
  auto gemm = [&](const u16* A, float* C, int M, int K, int Nc) {
    if (Nc % 64 == 0) {
      k_gemmN<4><<<dim3((M / 16) * (Nc / 64)), dim3(32), 0, stream>>>(A, wWt, C, M, K, Nc);
    } else {  // Nc == 48
      k_gemmN<3><<<dim3((M / 16) * (Nc / 48)), dim3(32), 0, stream>>>(A, wWt, C, M, K, Nc);
    }
  };

  // graph preprocessing
  k_graph_init<<<1, 256, 0, stream>>>(wDeg, wCnt, wCur);
  k_deg<<<(cE + cN + 255) / 256, 256, 0, stream>>>(ei, wDeg, wCnt);
  k_off_dinv<<<1, 256, 0, stream>>>(wCnt, wOff, wDeg, wDinv);
  k_fill<<<(cE + cN + 255) / 256, 256, 0, stream>>>(ei, wOff, wCur, wDinv, wColS, wColW);

  // input layernorm -> bf16 (rows of F=32)
  k_input_ln<<<cGN / 256, 256, 0, stream>>>(snap, normg, normb, wX0);

  // GCN block 1 (K=32)
  cvt(g1w, cF, cD); gemm(wX0, wHwG, cGN, cF, cD);
  cvt(r1w, cF, cD); gemm(wX0, wHwR, cGN, cF, cD);
  k_aggregate<<<cGN, 256, 0, stream>>>(wHwG, wHwR, g1b, r1b, wOff, wColS, wColW);
  k_ln256<<<cGN, 256, 0, stream>>>(wHwR, nullptr, nullptr, ln1g, ln1b, wX1, nullptr, 1);

  // GCN block 2
  cvt(g2w, cD, cD); gemm(wX1, wHwG, cGN, cD, cD);
  cvt(r2w, cD, cD); gemm(wX1, wHwR, cGN, cD, cD);
  k_aggregate<<<cGN, 256, 0, stream>>>(wHwG, wHwR, g2b, r2b, wOff, wColS, wColW);
  k_ln256<<<cGN, 256, 0, stream>>>(wHwR, nullptr, nullptr, ln2g, ln2b, wX0, nullptr, 1);

  // GCN block 3 (no activation)
  cvt(g3w, cD, cD); gemm(wX0, wHwG, cGN, cD, cD);
  cvt(r3w, cD, cD); gemm(wX0, wHwR, cGN, cD, cD);
  k_aggregate<<<cGN, 256, 0, stream>>>(wHwG, wHwR, g3b, r3b, wOff, wColS, wColW);
  k_ln256<<<cGN, 256, 0, stream>>>(wHwR, nullptr, nullptr, ln3g, ln3b, wX1, nullptr, 0);

  // pool over nodes -> emb
  k_pool<<<cG, 256, 0, stream>>>(wX1, wEmb, wEmbBf);

  // attention
  cvt(aiw, cD, 3 * cD); gemm(wEmbBf, wQkv, cG, cD, 3 * cD);
  k_attn<<<cB * cH, 64, 0, stream>>>(wQkv, aib, wAoBf);
  cvt(aow, cD, cD); gemm(wAoBf, wAoPr, cG, cD, cD);
  k_ln256<<<cG, 256, 0, stream>>>(wEmb, wAoPr, aob, alg, alb, wEmb2Bf, nullptr, 0);

  // mamba
  cvt(minw, cD, 2 * cDI); gemm(wEmb2Bf, wXz, cG, cD, 2 * cDI);
  k_conv<<<(cB * cDI + 255) / 256, 256, 0, stream>>>(wXz, mcw, mcb, wUs, wUsBf);
  cvt(mxw, cDI, 48); gemm(wUsBf, wXdb, cG, cDI, 48);
  k_dt<<<(cG * cDI + 255) / 256, 256, 0, stream>>>(wXdb, mdtw, mdtb, wDt);
  k_scan<<<(cB * cDI + 255) / 256, 256, 0, stream>>>(wDt, wUs, wXdb, wXz, mAlog, mDp, wYl);

  // head
  k_head<<<cB, 256, 0, stream>>>(wYl, mow, h1w, h1b, hl1g, hl1b, h2w, h2b, hl2g, hl2b, h3w, h3b, out);
}
